// WindowAttention_4801773437206
// MI455X (gfx1250) — matrix operand order repaired and checked by reading, not compiled, not run on hardware
//
#include <hip/hip_runtime.h>
#include <hip/hip_bf16.h>

typedef __attribute__((ext_vector_type(16))) _Float16 v16h;
typedef __attribute__((ext_vector_type(8)))  _Float16 v8h;
typedef __attribute__((ext_vector_type(8)))  float    v8f;
typedef __attribute__((ext_vector_type(4)))  int      v4i;

#define DIM     192
#define WS      8
#define HEADS   6
#define SHIFT   4
#define HD      32
#define WS2     64
#define HTOT    256
#define TOTWIN  4096          // B(4) * 32 * 32 windows
#define NTILE_QKV 9           // 576 / 64
#define NTILE_PRJ 3           // 192 / 64
#define SCALE_F 0.17677669529663689f   // 32^-0.5

// ---- optional CDNA5 async global->LDS path (guarded; falls back to plain copies)
#if defined(__has_builtin)
#if __has_builtin(__builtin_amdgcn_global_load_async_to_lds_b128) && \
    __has_builtin(__builtin_amdgcn_s_wait_asynccnt)
#define USE_ASYNC_LDS 1
#endif
#endif

#ifdef USE_ASYNC_LDS
// Toolchain signature (from rounds 1-2 diagnostics):
//   param0 = AS1 (global)  v4i*  (non-const)
//   param1 = AS3 (shared)  v4i*
//   param2/3 = int imm offset, int cpol
typedef __attribute__((address_space(1))) v4i* gv4i_p;
typedef __attribute__((address_space(3))) v4i* lds_v4i_p;
#endif

// ---------------------------------------------------------------------------
// WMMA fragment loaders, matching CDNA5 ISA VGPR layouts (CDNA5 ISA §7.12.2)
// A (16x32 f16): lane m<16 -> row m, K = {g*8..g*8+7} U {16+g*8..}, g = lane>>4
// B (32x16 f16): lane -> column lane&15, K = {g*8..g*8+7} U {16+g*8..}, g = lane>>4
// M is stored row-major in LDS with leading dim `ld` (f16 elements).
// For B we store "column n of B" as a contiguous LDS row (i.e. K/weights are
// kept as [n][k]), so the B fragment is two 16-byte ds loads per lane.
// ---------------------------------------------------------------------------
__device__ __forceinline__ v16h load_a_frag(const _Float16* M, int ld, int m0, int k0) {
  int lane = threadIdx.x & 31;
  int g = lane >> 4;
  const _Float16* p = M + (size_t)(m0 + (lane & 15)) * ld + k0 + g * 8;
  v8h lo = *(const v8h*)(p);
  v8h hi = *(const v8h*)(p + 16);
  v16h a;
#pragma unroll
  for (int i = 0; i < 8; ++i) { a[i] = lo[i]; a[i + 8] = hi[i]; }
  return a;
}

__device__ __forceinline__ v16h load_b_frag(const _Float16* M, int ld, int n0, int k0) {
  int lane = threadIdx.x & 31;
  int g = lane >> 4;
  const _Float16* p = M + (size_t)(n0 + (lane & 15)) * ld + k0 + g * 8;
  v8h lo = *(const v8h*)(p);
  v8h hi = *(const v8h*)(p + 16);
  v16h b;
#pragma unroll
  for (int i = 0; i < 8; ++i) { b[i] = lo[i]; b[i + 8] = hi[i]; }
  return b;
}

// ===========================================================================
// Kernel 1: fused roll(-4,-4) + im2row + QKV GEMM (64 tokens x 576) in f16.
// Output layout: [win][qkv(3)][head(6)][token(64)][hd(32)] f16
// Block: 256 threads = 8 waves (4 row-tiles x 2 col-tiles of 16x32).
// grid = (TOTWIN, 9): one window x one 64-wide N tile.
// ===========================================================================
__global__ __launch_bounds__(256)
void qkv_wmma_kernel(const float* __restrict__ x,
                     const float* __restrict__ qkv_w,
                     const float* __restrict__ qkv_b,
                     _Float16* __restrict__ qkv_out) {
  __shared__ alignas(32) _Float16 As[WS2 * DIM];   // tokens x channels
  __shared__ alignas(32) _Float16 Bs[64 * DIM];    // out-ch rows x in-ch
  const int win   = blockIdx.x;
  const int ntile = blockIdx.y;
  const int b  = win >> 10;
  const int wl = win & 1023;
  const int wh = wl >> 5, ww = wl & 31;
  const int tid = threadIdx.x;

  // --- stage A: gather one 8x8 window of the (-4,-4)-rolled NCHW image
  {
    const int t  = tid & 63;           // token
    const int q4 = tid >> 6;           // channel quarter
    const int i = t >> 3, j = t & 7;
    const int hs  = (wh * WS + i + SHIFT) & (HTOT - 1);
    const int ws_ = (ww * WS + j + SHIFT) & (HTOT - 1);
    const float* xp = x + (((size_t)b * DIM) * HTOT + hs) * HTOT + ws_;
#pragma unroll
    for (int cc = 0; cc < 48; ++cc) {
      const int c = q4 * 48 + cc;
      As[t * DIM + c] = (_Float16)xp[(size_t)c * HTOT * HTOT];
    }
  }
  // --- stage B: weight rows (column n of the GEMM-B as contiguous LDS row)
  {
    const int r  = tid & 63;
    const int q4 = tid >> 6;
    const float* wp = qkv_w + (size_t)(ntile * 64 + r) * DIM + q4 * 48;
    __builtin_prefetch(wp, 0, 1);      // global_prefetch_b8
#pragma unroll
    for (int cc = 0; cc < 48; ++cc)
      Bs[r * DIM + q4 * 48 + cc] = (_Float16)wp[cc];
  }
  __syncthreads();

  const int wv = tid >> 5;
  const int rowBase = (wv & 3) * 16;
  const int colBase = (wv >> 2) * 32;
  v8f acc0 = {}, acc1 = {};
#pragma unroll
  for (int k0 = 0; k0 < DIM; k0 += 32) {
    v16h a  = load_a_frag(As, DIM, rowBase, k0);
    v16h b0 = load_b_frag(Bs, DIM, colBase,      k0);
    v16h b1 = load_b_frag(Bs, DIM, colBase + 16, k0);
    acc0 = __builtin_amdgcn_wmma_f32_16x16x32_f16(false, a, false, b0, (short)0, acc0, false, false);
    acc1 = __builtin_amdgcn_wmma_f32_16x16x32_f16(false, a, false, b1, (short)0, acc1, false, false);
  }

  const int lane = tid & 31;
  const int hhalf = lane >> 4;
#pragma unroll
  for (int ct = 0; ct < 2; ++ct) {
    v8f acc = ct ? acc1 : acc0;
    const int col = colBase + ct * 16 + (lane & 15);
    const int o = ntile * 64 + col;            // 0..575
    const float bias = qkv_b[o];
    const int qi = o / DIM;
    const int rem = o - qi * DIM;
    const int head = rem >> 5, hd = rem & 31;
    _Float16* dst = qkv_out + ((((size_t)win * 3 + qi) * HEADS + head) * WS2) * HD + hd;
#pragma unroll
    for (int v = 0; v < 8; ++v) {
      const int tok = rowBase + v + 8 * hhalf;     // C layout: M = v + 8*(lane>=16)
      dst[(size_t)tok * HD] = (_Float16)(acc[v] + bias);
    }
  }
}

// ===========================================================================
// Kernel 2: one (window, head) per block. S = QK^T (WMMA, K=HD=32 exact),
// + scale + RPE + analytic shift mask, softmax via LDS, O = P*V (WMMA, K=64).
// Block: 128 threads = 4 waves; wave w owns S/O rows [16w, 16w+16).
// ===========================================================================
__global__ __launch_bounds__(128)
void attn_wmma_kernel(const _Float16* __restrict__ qkv,
                      const float* __restrict__ rpe_table,
                      _Float16* __restrict__ aout) {
  __shared__ alignas(32) _Float16 Qs[WS2 * HD];
  __shared__ alignas(32) _Float16 Ks[WS2 * HD];
  __shared__ alignas(32) _Float16 Vt[HD * WS2];    // V transposed: [hd][token]
  __shared__ float              Sx[WS2 * WS2];     // f32 scores
  __shared__ alignas(32) _Float16 Ps[WS2 * WS2];   // f16 probabilities
  __shared__ float              Rpe[(2 * WS - 1) * (2 * WS - 1)];
  __shared__ int                Reg[WS2];

  const int head = blockIdx.x % HEADS;
  const int win  = blockIdx.x / HEADS;
  const int wl = win & 1023;
  const int wh = wl >> 5, ww = wl & 31;
  const int tid = threadIdx.x;

  const _Float16* qg = qkv + (((size_t)win * 3 + 0) * HEADS + head) * (WS2 * HD);
  const _Float16* kg = qkv + (((size_t)win * 3 + 1) * HEADS + head) * (WS2 * HD);
  const _Float16* vg = qkv + (((size_t)win * 3 + 2) * HEADS + head) * (WS2 * HD);

#ifdef USE_ASYNC_LDS
  {
    v4i* qg4 = (v4i*)qg;               // drop const; builtin param is non-const
    v4i* kg4 = (v4i*)kg;
    v4i* qs4 = (v4i*)Qs;
    v4i* ks4 = (v4i*)Ks;
    for (int idx = tid; idx < (WS2 * HD) / 8; idx += 128) {
      __builtin_amdgcn_global_load_async_to_lds_b128(
          (gv4i_p)(qg4 + idx), (lds_v4i_p)(qs4 + idx), 0, 0);
      __builtin_amdgcn_global_load_async_to_lds_b128(
          (gv4i_p)(kg4 + idx), (lds_v4i_p)(ks4 + idx), 0, 0);
    }
  }
#else
  {
    const uint4* qg4 = (const uint4*)qg;
    const uint4* kg4 = (const uint4*)kg;
    uint4* qs4 = (uint4*)Qs;
    uint4* ks4 = (uint4*)Ks;
    for (int idx = tid; idx < (WS2 * HD) / 8; idx += 128) {
      qs4[idx] = qg4[idx];
      ks4[idx] = kg4[idx];
    }
  }
#endif
  // V staged transposed so the P*V B-fragment is a pair of 16B ds loads
  for (int idx = tid; idx < WS2 * HD; idx += 128) {
    const int t = idx >> 5, hd = idx & 31;
    Vt[hd * WS2 + t] = vg[idx];
  }
  // per-head RPE gather into LDS
  for (int idx = tid; idx < (2 * WS - 1) * (2 * WS - 1); idx += 128)
    Rpe[idx] = rpe_table[idx * HEADS + head];
  // analytic shift-mask region id per token (in rolled coordinates)
  if (tid < WS2) {
    const int ty = tid >> 3, tx = tid & 7;
    const int hh = wh * WS + ty, wp = ww * WS + tx;
    const int rh = (hh < HTOT - WS) ? 0 : (hh < HTOT - SHIFT) ? 1 : 2;
    const int rw = (wp < HTOT - WS) ? 0 : (wp < HTOT - SHIFT) ? 1 : 2;
    Reg[tid] = rh * 3 + rw;
  }
#ifdef USE_ASYNC_LDS
  __builtin_amdgcn_s_wait_asynccnt(0);
#endif
  __syncthreads();

  const int wv = tid >> 5;
  const int row0 = wv * 16;
  const int lane = tid & 31;
  const int hhalf = lane >> 4;

  // ---- S = Q K^T : 4 column tiles, K = HD = 32 -> one WMMA each
  {
    v16h a = load_a_frag(Qs, HD, row0, 0);
    v8f s0 = {}, s1 = {}, s2 = {}, s3 = {};
    {
      v16h b = load_b_frag(Ks, HD, 0, 0);
      s0 = __builtin_amdgcn_wmma_f32_16x16x32_f16(false, a, false, b, (short)0, s0, false, false);
    }
    {
      v16h b = load_b_frag(Ks, HD, 16, 0);
      s1 = __builtin_amdgcn_wmma_f32_16x16x32_f16(false, a, false, b, (short)0, s1, false, false);
    }
    {
      v16h b = load_b_frag(Ks, HD, 32, 0);
      s2 = __builtin_amdgcn_wmma_f32_16x16x32_f16(false, a, false, b, (short)0, s2, false, false);
    }
    {
      v16h b = load_b_frag(Ks, HD, 48, 0);
      s3 = __builtin_amdgcn_wmma_f32_16x16x32_f16(false, a, false, b, (short)0, s3, false, false);
    }
#pragma unroll
    for (int ct = 0; ct < 4; ++ct) {
      v8f s = (ct == 0) ? s0 : (ct == 1) ? s1 : (ct == 2) ? s2 : s3;
      const int col = ct * 16 + (lane & 15);
      const int ky = col >> 3, kx = col & 7;
#pragma unroll
      for (int v = 0; v < 8; ++v) {
        const int row = row0 + v + 8 * hhalf;
        const int qy = row >> 3, qx = row & 7;
        const float rp = Rpe[(qy - ky + 7) * 15 + (qx - kx + 7)];
        const float msk = (Reg[row] == Reg[col]) ? 0.0f : -1.0e30f;
        Sx[row * WS2 + col] = s[v] * SCALE_F + rp + msk;
      }
    }
  }
  __syncthreads();

  // ---- row softmax (one thread per row), re-emit as f16 P
  if (tid < WS2) {
    float mx = -3.0e38f;
    for (int c = 0; c < WS2; ++c) mx = fmaxf(mx, Sx[tid * WS2 + c]);
    float sum = 0.0f;
    for (int c = 0; c < WS2; ++c) {
      const float e = __expf(Sx[tid * WS2 + c] - mx);
      Sx[tid * WS2 + c] = e;
      sum += e;
    }
    const float inv = 1.0f / sum;
    for (int c = 0; c < WS2; ++c)
      Ps[tid * WS2 + c] = (_Float16)(Sx[tid * WS2 + c] * inv);
  }
  __syncthreads();

  // ---- O = P V : K = 64 -> two K steps, 2 column tiles (HD = 32)
  v8f o0 = {}, o1 = {};
#pragma unroll
  for (int k0 = 0; k0 < WS2; k0 += 32) {
    v16h a  = load_a_frag(Ps, WS2, row0, k0);
    v16h b0 = load_b_frag(Vt, WS2, 0,  k0);
    v16h b1 = load_b_frag(Vt, WS2, 16, k0);
    o0 = __builtin_amdgcn_wmma_f32_16x16x32_f16(false, a, false, b0, (short)0, o0, false, false);
    o1 = __builtin_amdgcn_wmma_f32_16x16x32_f16(false, a, false, b1, (short)0, o1, false, false);
  }
#pragma unroll
  for (int ct = 0; ct < 2; ++ct) {
    v8f oc = ct ? o1 : o0;
    const int ch = head * HD + ct * 16 + (lane & 15);
#pragma unroll
    for (int v = 0; v < 8; ++v) {
      const int tok = row0 + v + 8 * hhalf;
      aout[((size_t)win * WS2 + tok) * DIM + ch] = (_Float16)oc[v];
    }
  }
}

// ===========================================================================
// Kernel 3: output projection GEMM + bias, fused roll(+4,+4) and NHWC->NCHW
// scatter (roll round-trips each token to its original pixel).
// grid = (TOTWIN, 3); block 256 = 8 waves of 16x32 tiles.
// ===========================================================================
__global__ __launch_bounds__(256)
void proj_wmma_kernel(const _Float16* __restrict__ ain,
                      const float* __restrict__ proj_w,
                      const float* __restrict__ proj_b,
                      float* __restrict__ out) {
  __shared__ alignas(32) _Float16 As[WS2 * DIM];
  __shared__ alignas(32) _Float16 Bs[64 * DIM];
  const int win = blockIdx.x, ntile = blockIdx.y;
  const int b = win >> 10, wl = win & 1023;
  const int wh = wl >> 5, ww = wl & 31;
  const int tid = threadIdx.x;

  {
    const uint4* ag = (const uint4*)(ain + (size_t)win * WS2 * DIM);
    uint4* as4 = (uint4*)As;
    for (int idx = tid; idx < (WS2 * DIM) / 8; idx += 256) as4[idx] = ag[idx];
  }
  {
    const int r = tid & 63, q4 = tid >> 6;
    const float* wp = proj_w + (size_t)(ntile * 64 + r) * DIM + q4 * 48;
    __builtin_prefetch(wp, 0, 1);
#pragma unroll
    for (int cc = 0; cc < 48; ++cc)
      Bs[r * DIM + q4 * 48 + cc] = (_Float16)wp[cc];
  }
  __syncthreads();

  const int wv = tid >> 5;
  const int rowBase = (wv & 3) * 16;
  const int colBase = (wv >> 2) * 32;
  v8f acc0 = {}, acc1 = {};
#pragma unroll
  for (int k0 = 0; k0 < DIM; k0 += 32) {
    v16h a  = load_a_frag(As, DIM, rowBase, k0);
    v16h b0 = load_b_frag(Bs, DIM, colBase,      k0);
    v16h b1 = load_b_frag(Bs, DIM, colBase + 16, k0);
    acc0 = __builtin_amdgcn_wmma_f32_16x16x32_f16(false, a, false, b0, (short)0, acc0, false, false);
    acc1 = __builtin_amdgcn_wmma_f32_16x16x32_f16(false, a, false, b1, (short)0, acc1, false, false);
  }

  const int lane = tid & 31;
  const int hhalf = lane >> 4;
#pragma unroll
  for (int ct = 0; ct < 2; ++ct) {
    v8f acc = ct ? acc1 : acc0;
    const int col = colBase + ct * 16 + (lane & 15);
    const int ch = ntile * 64 + col;
    const float bias = proj_b[ch];
#pragma unroll
    for (int v = 0; v < 8; ++v) {
      const int tok = rowBase + v + 8 * hhalf;
      const int i = tok >> 3, j = tok & 7;
      const int hf = (wh * WS + i + SHIFT) & (HTOT - 1);
      const int wf = (ww * WS + j + SHIFT) & (HTOT - 1);
      out[(((size_t)b * DIM + ch) * HTOT + hf) * HTOT + wf] = acc[v] + bias;
    }
  }
}

// ===========================================================================
extern "C" void kernel_launch(void* const* d_in, const int* in_sizes, int n_in,
                              void* d_out, int out_size, void* d_ws, size_t ws_size,
                              hipStream_t stream) {
  (void)in_sizes; (void)n_in; (void)out_size; (void)ws_size;
  const float* x      = (const float*)d_in[0];
  const float* qkv_w  = (const float*)d_in[1];
  const float* qkv_b  = (const float*)d_in[2];
  const float* proj_w = (const float*)d_in[3];
  const float* proj_b = (const float*)d_in[4];
  const float* rpe    = (const float*)d_in[5];
  float* out = (float*)d_out;

  // workspace: qkv f16 [win][3][head][64][32], then attn-out f16 [win][64][192]
  _Float16* qkv_ws = (_Float16*)d_ws;
  const size_t qkvBytes = (size_t)TOTWIN * 3 * HEADS * WS2 * HD * sizeof(_Float16);
  _Float16* attn_ws = (_Float16*)((char*)d_ws + qkvBytes);

  qkv_wmma_kernel<<<dim3(TOTWIN, NTILE_QKV), 256, 0, stream>>>(x, qkv_w, qkv_b, qkv_ws);
  attn_wmma_kernel<<<dim3(TOTWIN * HEADS), 128, 0, stream>>>(qkv_ws, rpe, attn_ws);
  proj_wmma_kernel<<<dim3(TOTWIN, NTILE_PRJ), 256, 0, stream>>>(attn_ws, proj_w, proj_b, out);
}
